// MultiHeadSelfAttention_24232205484475
// MI455X (gfx1250) — matrix operand order repaired and checked by reading, not compiled, not run on hardware
//
#include <hip/hip_runtime.h>
#include <hip/hip_bf16.h>

// ---------------------------------------------------------------------------
// Multi-head self-attention for MI455X (gfx1250), all GEMMs via
// v_wmma_f32_16x16x32_bf16 (bf16 operands, fp32 accumulate).
//
// Shapes: B=2, T=2048, D=1024, H=16, DK=DV=64.
// d_out = [ output (B,T,D) fp32 | weights (B,H,T,T) fp32 ].
// ---------------------------------------------------------------------------

typedef __attribute__((ext_vector_type(16))) __bf16 v16bf;
typedef __attribute__((ext_vector_type(8)))  __bf16 v8bf;
typedef __attribute__((ext_vector_type(8)))  float  v8f;

static constexpr int kB  = 2;
static constexpr int kT  = 2048;
static constexpr int kD  = 1024;   // d_model
static constexpr int kH  = 16;
static constexpr int kDk = 64;
static constexpr int kDv = 64;

// ---------------------------------------------------------------------------
// WMMA helpers
// ---------------------------------------------------------------------------
__device__ __forceinline__ v8f wmma_bf16(v16bf a, v16bf b, v8f c) {
  // 8 args: (neg_a, A, neg_b, B, c_mod, C, reuse_a, reuse_b)
  return __builtin_amdgcn_wmma_f32_16x16x32_bf16(false, a, false, b,
                                                 (short)0, c, false, false);
}

// A-fragment (16x32 bf16). Lane holds row M = lane&15.
// elems 0..15 = K = 2*kbase8 + 0..15   (kbase8 = (lane>>4)*8)
__device__ __forceinline__ v16bf load_a32(const __bf16* __restrict__ row, int kbase8) {
  return *(const v16bf*)(row + 2 * kbase8);
}

// B-fragment (32x16 bf16). Lane holds col N = lane&15; 16 consecutive K values
// starting at (lane>>4)*16. Caller passes a pointer such that those 16 values
// are contiguous (32B aligned).
__device__ __forceinline__ v16bf load_b32(const __bf16* __restrict__ p) {
  return *(const v16bf*)p;
}

// ---------------------------------------------------------------------------
// fp32 -> bf16 conversion
// ---------------------------------------------------------------------------
__global__ void f32_to_bf16_kernel(const float* __restrict__ in,
                                   __bf16* __restrict__ out, int n) {
  int i = blockIdx.x * blockDim.x + threadIdx.x;
  if (i < n) out[i] = (__bf16)in[i];
}

// ---------------------------------------------------------------------------
// QKV projection: for each (b,h): Q = x (T x D) * W^T (D x 64).
// W kept in original (H, 64, D) row-major layout -> B[kk][n] = W[h][n][kk]
// is contiguous in kk. One workgroup (4 waves) = one 16x64 tile of one matrix.
// grid = (B*T/16, 3*H); blockIdx.y: mat = y/H (0=Q,1=K,2=V), h = y%H.
// V is stored transposed: Vt[b][h][v][t].
// ---------------------------------------------------------------------------
__global__ __launch_bounds__(128)
void qkv_proj_kernel(const __bf16* __restrict__ xb,
                     const __bf16* __restrict__ wqb,
                     const __bf16* __restrict__ wkb,
                     const __bf16* __restrict__ wvb,
                     __bf16* __restrict__ Qb,
                     __bf16* __restrict__ Kb,
                     __bf16* __restrict__ Vtb) {
  const int lane = threadIdx.x & 31;
  const int wave = threadIdx.x >> 5;       // N-tile 0..3
  const int rowTile = blockIdx.x;          // tiles over B*T
  const int mat = blockIdx.y / kH;         // 0=Q,1=K,2=V
  const int h   = blockIdx.y % kH;

  const __bf16* W = (mat == 0) ? wqb : (mat == 1) ? wkb : wvb;
  W += (size_t)h * kDk * kD;

  const int arow = rowTile * 16 + (lane & 15);
  const int kb8  = (lane >> 4) * 8;
  const int kb16 = (lane >> 4) * 16;
  const int ncol = wave * 16 + (lane & 15);

  const __bf16* xrow = xb + (size_t)arow * kD;
  const __bf16* wrow = W  + (size_t)ncol * kD;

  v8f acc = {};
#pragma unroll 4
  for (int kk = 0; kk < kD; kk += 32) {
    acc = wmma_bf16(load_a32(xrow + kk, kb8), load_b32(wrow + kk + kb16), acc);
  }

  // Destination pointer + stride hoisted (a 16-row tile never crosses batch).
  const int m0 = rowTile * 16 + (lane >> 4) * 8;   // global row of acc[0]
  const int bb = m0 / kT, t0 = m0 % kT;
  __bf16* dst;
  int stride;
  if (mat == 0) {
    dst = Qb + (((size_t)bb * kH + h) * kT + t0) * kDk + ncol;  stride = kDk;
  } else if (mat == 1) {
    dst = Kb + (((size_t)bb * kH + h) * kT + t0) * kDk + ncol;  stride = kDk;
  } else {
    dst = Vtb + (((size_t)bb * kH + h) * kDv + ncol) * kT + t0; stride = 1;
  }
#pragma unroll
  for (int i = 0; i < 8; ++i) dst[(size_t)i * stride] = (__bf16)acc[i];
}

// ---------------------------------------------------------------------------
// Attention: one workgroup (8 waves, 256 thr) per (b, h, 16-row tile of T).
// Phase A: scores (16 x 2048) = Q-tile * K^T via WMMA -> LDS (fp32, 128 KB).
// Phase B: masked softmax over the strip; normalized weights written to HBM.
// Phase C: out-tile (16 x 64) = P * V via WMMA (split-K across wave pairs).
// ---------------------------------------------------------------------------
__global__ __launch_bounds__(256)
void attn_kernel(const __bf16* __restrict__ Qb,
                 const __bf16* __restrict__ Kb,
                 const __bf16* __restrict__ Vtb,
                 const unsigned char* __restrict__ mask,  // (B,T) bool
                 float* __restrict__ weightsOut,          // (B,H,T,T)
                 __bf16* __restrict__ concat) {           // (B,T,H*DV)
  __shared__ float s_scores[16 * kT];       // 128 KB (CDNA5 320KB LDS/WGP)
  __shared__ float s_red[16 * 16];
  __shared__ float s_rowmax[16];
  __shared__ float s_inv[16];
  __shared__ float s_part[2 * 16 * kDv];    // split-K partials

  const int tid  = threadIdx.x;
  const int lane = tid & 31;
  const int wave = tid >> 5;                // 0..7
  const int mt = blockIdx.x % (kT / 16);
  const int h  = (blockIdx.x / (kT / 16)) % kH;
  const int bb = blockIdx.x / ((kT / 16) * kH);

  const size_t headBase = ((size_t)bb * kH + h) * (size_t)kT * kDk;
  const int kb8  = (lane >> 4) * 8;
  const int kb16 = (lane >> 4) * 16;

  // Q A-fragments for this row tile (16 x 64 = two k-steps), loaded once.
  const __bf16* Qrow = Qb + headBase + (size_t)(mt * 16 + (lane & 15)) * kDk;
  const v16bf qa0 = load_a32(Qrow + 0,  kb8);
  const v16bf qa1 = load_a32(Qrow + 32, kb8);

  const float scale = 0.125f;               // 1/sqrt(64)

  // ---- Phase A: scores -> LDS ----
  const int colsPerWave = kT / 8;           // 256
#pragma unroll 4
  for (int ct = 0; ct < colsPerWave / 16; ++ct) {
    const int s = wave * colsPerWave + ct * 16 + (lane & 15);
    const __bf16* krow = Kb + headBase + (size_t)s * kDk;
    if (ct + 1 < colsPerWave / 16)
      __builtin_prefetch(krow + 16 * kDk, 0, 0);   // next tile's K rows
    v8f acc = {};
    acc = wmma_bf16(qa0, load_b32(krow + 0  + kb16), acc);
    acc = wmma_bf16(qa1, load_b32(krow + 32 + kb16), acc);
    const bool pad = mask[(size_t)bb * kT + s] != 0;
    const int m0 = (lane >> 4) * 8;
#pragma unroll
    for (int i = 0; i < 8; ++i) {
      s_scores[(m0 + i) * kT + s] = pad ? -__builtin_inff() : acc[i] * scale;
    }
  }
  __syncthreads();

  // ---- Phase B: softmax (row max, exp, sum, normalize) ----
  const int r = tid & 15;
  const int cp = tid >> 4;                  // 0..15
  float m = -__builtin_inff();
  for (int c = cp; c < kT; c += 16) m = fmaxf(m, s_scores[r * kT + c]);
  s_red[r * 16 + cp] = m;
  __syncthreads();
  if (tid < 16) {
    float mm = -__builtin_inff();
#pragma unroll
    for (int j = 0; j < 16; ++j) mm = fmaxf(mm, s_red[tid * 16 + j]);
    s_rowmax[tid] = mm;
  }
  __syncthreads();
  const float rm = s_rowmax[r];
  float sum = 0.f;
  for (int c = cp; c < kT; c += 16) {
    const float v = s_scores[r * kT + c];
    const float p = (rm == -__builtin_inff()) ? 0.f : __expf(v - rm);
    s_scores[r * kT + c] = p;
    sum += p;
  }
  s_red[r * 16 + cp] = sum;
  __syncthreads();
  if (tid < 16) {
    float ss = 0.f;
#pragma unroll
    for (int j = 0; j < 16; ++j) ss += s_red[tid * 16 + j];
    s_inv[tid] = (ss > 0.f) ? 1.f / ss : 0.f;   // nan_to_num for masked rows
  }
  __syncthreads();
  // normalize in LDS + coalesced weights write to HBM
  float* wout = weightsOut + (((size_t)bb * kH + h) * kT + (size_t)mt * 16) * kT;
  for (int idx = tid; idx < 16 * kT; idx += 256) {
    const float p = s_scores[idx] * s_inv[idx >> 11];   // idx/2048 = row
    s_scores[idx] = p;
    wout[idx] = p;
  }
  __syncthreads();

  // ---- Phase C: out = P (16 x 2048) * V (2048 x 64), split-K over wave pairs
  const int ct   = wave & 3;                // V col-tile
  const int kh   = wave >> 2;               // K half
  const int vcol = ct * 16 + (lane & 15);
  const __bf16* vrow = Vtb + ((size_t)bb * kH + h) * (size_t)kDv * kT
                           + (size_t)vcol * kT;
  const int arow = lane & 15;
  v8f acc = {};
  const int k0beg = kh * (kT / 2), k0end = k0beg + kT / 2;
#pragma unroll 4
  for (int k0 = k0beg; k0 < k0end; k0 += 32) {
    if (k0 + 128 < k0end)
      __builtin_prefetch(vrow + k0 + 128 + kb16, 0, 0);  // upcoming V rows
    v16bf a;
    const float* sp = s_scores + arow * kT + k0 + 2 * kb8;
#pragma unroll
    for (int j = 0; j < 8; ++j) { a[j] = (__bf16)sp[j]; a[j + 8] = (__bf16)sp[j + 8]; }
    acc = wmma_bf16(a, load_b32(vrow + k0 + kb16), acc);
  }
  const int m0 = (lane >> 4) * 8;
#pragma unroll
  for (int i = 0; i < 8; ++i)
    s_part[kh * (16 * kDv) + (m0 + i) * kDv + vcol] = acc[i];
  __syncthreads();

  __bf16* crow = concat + ((size_t)bb * kT + (size_t)mt * 16) * (kH * kDv) + h * kDv;
  for (int idx = tid; idx < 16 * kDv; idx += 256) {
    const int rr = idx >> 6, cc = idx & 63;
    const float v = s_part[idx] + s_part[16 * kDv + idx];
    crow[(size_t)rr * (kH * kDv) + cc] = (__bf16)v;
  }
}

// ---------------------------------------------------------------------------
// Output projection: out (B*T x D) = concat (B*T x 1024) * W_O^T.
// W_O kept in original (D, H*DV) row-major -> B[kk][n] = W_O[n][kk] contiguous.
// grid = (B*T/16, D/64); 4 waves per workgroup, one 16x16 tile each.
// ---------------------------------------------------------------------------
__global__ __launch_bounds__(128)
void out_proj_kernel(const __bf16* __restrict__ cb,
                     const __bf16* __restrict__ wob,
                     float* __restrict__ outD) {
  const int lane = threadIdx.x & 31;
  const int wave = threadIdx.x >> 5;
  const int rowTile = blockIdx.x;
  const int ncol = blockIdx.y * 64 + wave * 16 + (lane & 15);
  const int kb8  = (lane >> 4) * 8;
  const int kb16 = (lane >> 4) * 16;

  const __bf16* arow = cb  + (size_t)(rowTile * 16 + (lane & 15)) * kD;
  const __bf16* brow = wob + (size_t)ncol * kD;

  v8f acc = {};
#pragma unroll 4
  for (int kk = 0; kk < kD; kk += 32)
    acc = wmma_bf16(load_a32(arow + kk, kb8), load_b32(brow + kk + kb16), acc);

  const int m0 = rowTile * 16 + (lane >> 4) * 8;
#pragma unroll
  for (int i = 0; i < 8; ++i)
    outD[(size_t)(m0 + i) * kD + ncol] = acc[i];
}

// ---------------------------------------------------------------------------
// Launch
// ---------------------------------------------------------------------------
extern "C" void kernel_launch(void* const* d_in, const int* in_sizes, int n_in,
                              void* d_out, int out_size, void* d_ws, size_t ws_size,
                              hipStream_t stream) {
  (void)in_sizes; (void)n_in; (void)out_size; (void)ws_size;

  const float* x   = (const float*)d_in[0];
  const unsigned char* mask = (const unsigned char*)d_in[1];   // bool (B,1,1,T)
  const float* W_Q = (const float*)d_in[2];
  const float* W_K = (const float*)d_in[3];
  const float* W_V = (const float*)d_in[4];
  const float* W_O = (const float*)d_in[5];

  float* outD = (float*)d_out;                          // (B,T,D)
  float* weightsOut = outD + (size_t)kB * kT * kD;      // (B,H,T,T)

  // Workspace carving (256B aligned). Total ~48 MB.
  char* ws = (char*)d_ws;
  size_t off = 0;
  auto carve = [&](size_t bytes) {
    char* p = ws + off;
    off += (bytes + 255) & ~(size_t)255;
    return p;
  };
  const size_t nX  = (size_t)kB * kT * kD;        // 4,194,304
  const size_t nW  = (size_t)kH * kDk * kD;       // 1,048,576
  const size_t nWo = (size_t)kD * (kH * kDv);     // 1,048,576
  const size_t nQ  = (size_t)kB * kH * kT * kDk;  // 4,194,304

  __bf16* xb  = (__bf16*)carve(nX  * 2);
  __bf16* wqb = (__bf16*)carve(nW  * 2);
  __bf16* wkb = (__bf16*)carve(nW  * 2);
  __bf16* wvb = (__bf16*)carve(nW  * 2);
  __bf16* wob = (__bf16*)carve(nWo * 2);
  __bf16* Qb  = (__bf16*)carve(nQ  * 2);
  __bf16* Kb  = (__bf16*)carve(nQ  * 2);
  __bf16* Vtb = (__bf16*)carve(nQ  * 2);
  __bf16* cat = (__bf16*)carve(nX  * 2);

  // 1) fp32 -> bf16 conversions
  {
    const int blk = 256;
    f32_to_bf16_kernel<<<(int)((nX  + blk - 1) / blk), blk, 0, stream>>>(x,   xb,  (int)nX);
    f32_to_bf16_kernel<<<(int)((nW  + blk - 1) / blk), blk, 0, stream>>>(W_Q, wqb, (int)nW);
    f32_to_bf16_kernel<<<(int)((nW  + blk - 1) / blk), blk, 0, stream>>>(W_K, wkb, (int)nW);
    f32_to_bf16_kernel<<<(int)((nW  + blk - 1) / blk), blk, 0, stream>>>(W_V, wvb, (int)nW);
    f32_to_bf16_kernel<<<(int)((nWo + blk - 1) / blk), blk, 0, stream>>>(W_O, wob, (int)nWo);
  }

  // 2) QKV projections (WMMA)
  {
    dim3 grid(kB * kT / 16, 3 * kH);
    qkv_proj_kernel<<<grid, 128, 0, stream>>>(xb, wqb, wkb, wvb, Qb, Kb, Vtb);
  }

  // 3) attention + softmax + weights write + PV (WMMA)
  {
    dim3 grid(kB * kH * (kT / 16));
    attn_kernel<<<grid, 256, 0, stream>>>(Qb, Kb, Vtb, mask, weightsOut, cat);
  }

  // 4) output projection (WMMA)
  {
    dim3 grid(kB * kT / 16, kD / 64);
    out_proj_kernel<<<grid, 128, 0, stream>>>(cat, wob, outD);
  }
}
